// Discriminator_67053029425159
// MI455X (gfx1250) — matrix operand order repaired and checked by reading, compiled, not run on hardware
//
#include <hip/hip_runtime.h>
#include <stdint.h>

#define BATCH   1024
#define EMBED   16
#define HIDDEN  256
#define H4      1024
#define H2      512
#define NEXP    7
#define S_MAX   16641
#define D_IN    (S_MAX + EMBED)

typedef __attribute__((ext_vector_type(16))) __bf16 v16bf;
typedef __attribute__((ext_vector_type(8)))  __bf16 v8bf;
typedef __attribute__((ext_vector_type(8)))  float  v8f;
typedef __attribute__((ext_vector_type(4)))  float  v4f;

__constant__ int c_sizes[NEXP] = {9, 25, 81, 289, 1089, 4225, 16641};

static __device__ __forceinline__ __bf16 f2bf(float f) { return (__bf16)f; }

// ---------------------------------------------------------------------------
// Kernel 1: group batch rows by expert (counts + gathered index lists).
// ---------------------------------------------------------------------------
__global__ void k_build_index(const int* __restrict__ orders,
                              int* __restrict__ counts, int* __restrict__ idx) {
    int t = threadIdx.x;
    if (t < NEXP) counts[t] = 0;
    __syncthreads();
    int o = orders[t];
    int pos = atomicAdd(&counts[o], 1);
    idx[o * BATCH + pos] = t;
}

// ---------------------------------------------------------------------------
// Kernel 2: expert input layer.  Block = 16 gathered rows x 128 cols, 8 waves,
// each wave one 16x16 WMMA tile.  Fast path (all 32 K-rows inside the maze
// slice): coalesced float4 weight staging -> bf16 LDS [N][K]; fragments are
// aligned ds_load_b128 reads.  Tail path handles the maze/embed boundary.
// ---------------------------------------------------------------------------
__global__ void k_input_layer(const float* __restrict__ mazes,
                              const float* __restrict__ embed,
                              const float* __restrict__ W_in,
                              const float* __restrict__ b_in,
                              const int* __restrict__ counts,
                              const int* __restrict__ idx,
                              float* __restrict__ h) {
    __shared__ __align__(32) __bf16 ldsA[16 * 32];     // [m][k]
    __shared__ __align__(32) __bf16 ldsB[128 * 32];    // [n][k]
    __shared__ int ldsIdx[16];

    const int o   = blockIdx.z;
    const int cnt = counts[o];
    const int rt  = blockIdx.y;
    if (rt * 16 >= cnt) return;

    const int s    = c_sizes[o];
    const int Ktot = s + EMBED;
    const int tid  = threadIdx.x;
    const int lane = tid & 31;
    const int wave = tid >> 5;
    const int half = lane >> 4;
    const int nl   = lane & 15;
    const int m    = lane & 15;
    const int nb0  = blockIdx.x * 128;
    const int col  = nb0 + wave * 16 + nl;

    if (tid < 16) {
        int g = rt * 16 + tid;
        ldsIdx[tid] = (g < cnt) ? idx[o * BATCH + g] : -1;
    }
    __syncthreads();

    // Loop-invariant staging address: this thread always stages row tid>>4.
    const int  bi0   = ldsIdx[tid >> 4];
    const long abase = (long)max(bi0, 0) * S_MAX;   // clamped; invalid rows never stored
    const int  k2    = (tid & 15) * 2;
    const float* arow = mazes + abase + k2;

    const float* Wo = W_in + (long)o * D_IN * H4;
    v8f acc = {};

    const int kfast = s & ~31;              // full-maze K steps

    // ---------------- fast path ----------------
    for (int kb = 0; kb < kfast; kb += 32) {
        __syncthreads();
        // A tile: 16 rows x 32 k, 2 consecutive k per thread (merges to b64).
        {
            int r = tid >> 4;
            ldsA[r * 32 + k2]     = f2bf(arow[kb]);
            ldsA[r * 32 + k2 + 1] = f2bf(arow[kb + 1]);
        }
        // B tile: 32 k x 128 n, float4 per thread x4, store [n][k] bf16.
        #pragma unroll
        for (int i = 0; i < 4; ++i) {
            int l  = tid + i * 256;
            int k  = l >> 5;
            int n4 = (l & 31) * 4;
            v4f w = *(const v4f*)&Wo[(long)(kb + k) * H4 + nb0 + n4];
            #pragma unroll
            for (int j = 0; j < 4; ++j)
                ldsB[(n4 + j) * 32 + k] = f2bf(w[j]);
        }
        if (kb + 32 < kfast)
            __builtin_prefetch(&Wo[(long)(kb + 32) * H4 + nb0], 0, 1);
        __syncthreads();

        v8bf  a0 = *(const v8bf*)&ldsA[m * 32 + half * 16];
        v8bf  a1 = *(const v8bf*)&ldsA[m * 32 + half * 16 + 8];
        v16bf af = __builtin_shufflevector(a0, a1, 0,1,2,3,4,5,6,7,8,9,10,11,12,13,14,15);
        v16bf wf = *(const v16bf*)&ldsB[(wave * 16 + nl) * 32 + half * 16];
        acc = __builtin_amdgcn_wmma_f32_16x16x32_bf16(
            false, af, false, wf, (short)0, acc, false, false);
    }

    // ---------------- tail path (boundary + embed rows) ----------------
    for (int kb = kfast; kb < Ktot; kb += 32) {
        __syncthreads();
        #pragma unroll
        for (int i = 0; i < 2; ++i) {
            int r  = tid >> 4;
            int kg = kb + k2 + i;
            float v = 0.f;
            if (bi0 >= 0 && kg < Ktot)
                v = (kg < s) ? mazes[abase + kg]
                             : embed[o * EMBED + (kg - s)];
            ldsA[r * 32 + k2 + i] = f2bf(v);
        }
        __syncthreads();

        v8bf  a0 = *(const v8bf*)&ldsA[m * 32 + half * 16];
        v8bf  a1 = *(const v8bf*)&ldsA[m * 32 + half * 16 + 8];
        v16bf af = __builtin_shufflevector(a0, a1, 0,1,2,3,4,5,6,7,8,9,10,11,12,13,14,15);
        v16bf wf;
        #pragma unroll
        for (int j = 0; j < 16; ++j) {
            int kg = kb + half * 16 + j;
            float wv = 0.f;
            if (kg < Ktot) {
                int rrow = (kg < s) ? kg : (S_MAX + (kg - s));
                wv = Wo[(long)rrow * H4 + col];
            }
            wf[j] = f2bf(wv);
        }
        acc = __builtin_amdgcn_wmma_f32_16x16x32_bf16(
            false, af, false, wf, (short)0, acc, false, false);
    }

    // Epilogue: bias + leaky_relu(0.2), scatter to dense h by batch row.
    float bias = b_in[o * H4 + col];
    #pragma unroll
    for (int v = 0; v < 8; ++v) {
        int mm = half * 8 + v;
        int g  = rt * 16 + mm;
        if (g < cnt) {
            float y = acc[v] + bias;
            y = fmaxf(y, 0.2f * y);
            h[(long)ldsIdx[mm] * H4 + col] = y;
        }
    }
}

// ---------------------------------------------------------------------------
// Kernel 3: dense GEMM  C = leaky(A[M,K] @ W[K,N] + bias), bf16 WMMA.
// Block tile 32x64, 8 waves (2M x 4N).  A tile staged f32 via
// global_load_async_to_lds_b128 (ASYNCcnt); W tile staged bf16 [N][K].
// ---------------------------------------------------------------------------
__global__ void k_gemm_bias_lrelu(const float* __restrict__ A,
                                  const float* __restrict__ W,
                                  const float* __restrict__ bias,
                                  float* __restrict__ C,
                                  int K, int N) {
    __shared__ __align__(16) float  ldsAf[32 * 32];    // [m][k] f32 (async dst)
    __shared__ __align__(32) __bf16 ldsB[64 * 32];     // [n][k] bf16

    const int tid  = threadIdx.x;
    const int lane = tid & 31;
    const int wave = tid >> 5;
    const int half = lane >> 4;
    const int nl   = lane & 15;
    const int mt   = wave >> 2;      // 0..1
    const int nt   = wave & 3;       // 0..3
    const int rowbase = blockIdx.y * 32;
    const int colbase = blockIdx.x * 64;
    const int m = mt * 16 + (lane & 15);

    // async A staging: each thread moves one 16B chunk per K step.
    const int ar  = tid >> 3;        // 0..31 row in tile
    const int ac4 = (tid & 7) * 4;   // 0..28 col (float) in tile
    const unsigned ldsoff = (unsigned)(uintptr_t)(&ldsAf[tid * 4]);
    const float* agp = A + (long)(rowbase + ar) * K + ac4;

    v8f acc = {};
    for (int kb = 0; kb < K; kb += 32) {
        __syncthreads();
        // A tile 32x32 f32: global -> LDS, no VGPR round trip.
        asm volatile("global_load_async_to_lds_b128 %0, %1, off"
                     :: "v"(ldsoff), "v"(agp + kb) : "memory");
        // W tile 32k x 64n: coalesced float4 loads -> bf16 LDS [n][k].
        #pragma unroll
        for (int i = 0; i < 2; ++i) {
            int l  = tid + i * 256;
            int k  = l >> 4;
            int n4 = (l & 15) * 4;
            v4f w = *(const v4f*)&W[(long)(kb + k) * N + colbase + n4];
            #pragma unroll
            for (int j = 0; j < 4; ++j)
                ldsB[(n4 + j) * 32 + k] = f2bf(w[j]);
        }
        if (kb + 32 < K)
            __builtin_prefetch(&W[(long)(kb + 32) * N + colbase], 0, 1);
        asm volatile("s_wait_asynccnt 0x0" ::: "memory");
        __syncthreads();

        v4f a0 = *(const v4f*)&ldsAf[m * 32 + half * 16];
        v4f a1 = *(const v4f*)&ldsAf[m * 32 + half * 16 + 4];
        v4f a2 = *(const v4f*)&ldsAf[m * 32 + half * 16 + 8];
        v4f a3 = *(const v4f*)&ldsAf[m * 32 + half * 16 + 12];
        v16bf af;
        #pragma unroll
        for (int j = 0; j < 4; ++j) {
            af[j]      = f2bf(a0[j]);
            af[j + 4]  = f2bf(a1[j]);
            af[j + 8]  = f2bf(a2[j]);
            af[j + 12] = f2bf(a3[j]);
        }
        v16bf wf = *(const v16bf*)&ldsB[(nt * 16 + nl) * 32 + half * 16];
        acc = __builtin_amdgcn_wmma_f32_16x16x32_bf16(
            false, af, false, wf, (short)0, acc, false, false);
    }

    float b = bias[colbase + nt * 16 + nl];
    #pragma unroll
    for (int v = 0; v < 8; ++v) {
        float y = acc[v] + b;
        y = fmaxf(y, 0.2f * y);
        C[(long)(rowbase + mt * 16 + half * 8 + v) * N + colbase + nt * 16 + nl] = y;
    }
}

// ---------------------------------------------------------------------------
// Kernel 4: final 256->1 dot + sigmoid.  One wave per batch row.
// ---------------------------------------------------------------------------
__global__ void k_final(const float* __restrict__ z2,
                        const float* __restrict__ W3,
                        const float* __restrict__ b3,
                        float* __restrict__ out) {
    const int tid  = threadIdx.x;
    const int lane = tid & 31;
    const int wave = tid >> 5;
    const int row  = blockIdx.x * 8 + wave;

    float x = 0.f;
    #pragma unroll
    for (int i = 0; i < 8; ++i) {
        int k = lane + i * 32;
        x += z2[(long)row * HIDDEN + k] * W3[k];
    }
    #pragma unroll
    for (int off = 16; off > 0; off >>= 1)
        x += __shfl_xor(x, off, 32);
    if (lane == 0)
        out[row] = 1.f / (1.f + __expf(-(x + b3[0])));
}

// ---------------------------------------------------------------------------
extern "C" void kernel_launch(void* const* d_in, const int* in_sizes, int n_in,
                              void* d_out, int out_size, void* d_ws, size_t ws_size,
                              hipStream_t stream) {
    const float* mazes  = (const float*)d_in[0];
    const int*   orders = (const int*)  d_in[1];
    const float* embed  = (const float*)d_in[2];
    const float* W_in   = (const float*)d_in[3];
    const float* b_in   = (const float*)d_in[4];
    const float* W1     = (const float*)d_in[5];
    const float* b1     = (const float*)d_in[6];
    const float* W2     = (const float*)d_in[7];
    const float* b2     = (const float*)d_in[8];
    const float* W3     = (const float*)d_in[9];
    const float* b3     = (const float*)d_in[10];
    float* out = (float*)d_out;

    char* ws = (char*)d_ws;
    int*   counts = (int*)ws;                   // 28 B
    int*   idx    = (int*)(ws + 1024);          // 28 KB
    float* h      = (float*)(ws + 32768);       // 1024x1024 f32 = 4 MB
    float* z1     = h  + (size_t)BATCH * H4;    // 1024x512  f32 = 2 MB
    float* z2     = z1 + (size_t)BATCH * H2;    // 1024x256  f32 = 1 MB

    k_build_index<<<1, 1024, 0, stream>>>(orders, counts, idx);
    k_input_layer<<<dim3(H4 / 128, BATCH / 16, NEXP), 256, 0, stream>>>(
        mazes, embed, W_in, b_in, counts, idx, h);
    k_gemm_bias_lrelu<<<dim3(H2 / 64, BATCH / 32), 256, 0, stream>>>(h,  W1, b1, z1, H4, H2);
    k_gemm_bias_lrelu<<<dim3(HIDDEN / 64, BATCH / 32), 256, 0, stream>>>(z1, W2, b2, z2, H2, HIDDEN);
    k_final<<<BATCH / 8, 256, 0, stream>>>(z2, W3, b3, out);
}
